// Lfm2Attention_3753801417574
// MI455X (gfx1250) — matrix operand order repaired and checked by reading, not compiled, not run on hardware
//
#include <hip/hip_runtime.h>

// Problem constants (fixed by the reference)
#define T_SEQ  2048
#define BATCH  2
#define DMODEL 1536
#define NH     24
#define NKV    8
#define HD     64
#define KVD    (NKV*HD)     // 512
#define GRP    (NH/NKV)     // 3
#define ROWS   (BATCH*T_SEQ)// 4096
#define EPS    1e-5f
#define SL2E   (0.125f * 1.44269504088896f)  // scale * log2(e), scale = hd^-0.5

typedef __bf16 bf16;
typedef __attribute__((ext_vector_type(16))) __bf16 bf16x16;
typedef __attribute__((ext_vector_type(8)))  __bf16 bf16x8;
typedef __attribute__((ext_vector_type(8)))  float  f32x8;
typedef __attribute__((ext_vector_type(4)))  unsigned int u32x4;
typedef __attribute__((ext_vector_type(8)))  int i32x8;
typedef __attribute__((ext_vector_type(4)))  int i32x4;

union Frag { bf16x16 v; bf16x8 h[2]; };

__device__ __forceinline__ f32x8 wmma_bf16(const Frag& a, const Frag& b, f32x8 c) {
  return __builtin_amdgcn_wmma_f32_16x16x32_bf16(false, a.v, false, b.v, (short)0, c,
                                                 false, false);
}

// ds_swizzle XOR exchange within 16-lane halves (group-of-32, and=0x1f)
template <int IMM>
__device__ __forceinline__ float swz(float v) {
  return __int_as_float(__builtin_amdgcn_ds_swizzle(__float_as_int(v), IMM));
}
__device__ __forceinline__ float redmax16(float v) {
  v = fmaxf(v, swz<0x041F>(v));  // xor 1
  v = fmaxf(v, swz<0x081F>(v));  // xor 2
  v = fmaxf(v, swz<0x101F>(v));  // xor 4
  v = fmaxf(v, swz<0x201F>(v));  // xor 8
  return v;
}
__device__ __forceinline__ float redsum16(float v) {
  v += swz<0x041F>(v); v += swz<0x081F>(v);
  v += swz<0x101F>(v); v += swz<0x201F>(v);
  return v;
}

// ---------------------------------------------------------------- TDM loader
// 2D tile (32 cols x `rows` rows of bf16) from a row-major [rows, rowstride]
// tensor into LDS, padding 4 DWORDs after every 16 DWORDs (64B row) so the
// LDS image has an 80-byte (40-element) row stride.
__device__ __forceinline__ void tdm_load_2d(const bf16* gptr, unsigned lds_off,
                                            int rows, int rowstride_elems) {
  const unsigned long long ga = (unsigned long long)(uintptr_t)gptr;
  u32x4 g0;
  g0[0] = 1u;                                  // count=1, user descriptor
  g0[1] = lds_off;                             // lds_addr
  g0[2] = (unsigned)ga;                        // global_addr[31:0]
  g0[3] = (unsigned)(ga >> 32) | (2u << 30);   // global_addr[56:32] | type=2
  i32x8 g1;
  g1[0] = (1 << 16)      // data_size = 2 bytes
        | (1 << 20)      // pad_enable
        | (3 << 22)      // pad_interval: 16 DWORDs (one 64B tile row)
        | (3 << 25);     // pad_amount: 4 DWORDs (16B) -> 40-elem LDS stride
  g1[1] = (32 & 0xFFFF) << 16;                 // tensor_dim0 = 32 (lo16)
  g1[2] = (rows & 0xFFFF) << 16;               // dim0 hi=0 | tensor_dim1 lo16
  g1[3] = ((rows >> 16) & 0xFFFF) | (32 << 16);// dim1 hi | tile_dim0 = 32
  g1[4] = rows & 0xFFFF;                       // tile_dim1 = rows, tile_dim2=0
  g1[5] = rowstride_elems;                     // tensor_dim0_stride lo32
  g1[6] = 0;                                   // stride0 hi | stride1 lo
  g1[7] = 0;
  const i32x4 gz = {0, 0, 0, 0};
#if defined(__clang_major__) && (__clang_major__ >= 23)
  const i32x8 gz8 = {0, 0, 0, 0, 0, 0, 0, 0};
  __builtin_amdgcn_tensor_load_to_lds(g0, g1, gz, gz, gz8, 0);
#else
  __builtin_amdgcn_tensor_load_to_lds(g0, g1, gz, gz, 0);
#endif
}
__device__ __forceinline__ unsigned lds_of(const void* p) {
  return (unsigned)(unsigned long long)(uintptr_t)p;  // low 32 bits = LDS offset
}

// ---------------------------------------------------------------- fp32 -> bf16
__global__ __launch_bounds__(256) void cvt_kernel(const float* __restrict__ in,
                                                  bf16* __restrict__ out, int n4) {
  int i = blockIdx.x * blockDim.x + threadIdx.x;
  if (i < n4) {
    float4 v = reinterpret_cast<const float4*>(in)[i];
    bf16* o = out + i * 4;
    o[0] = (bf16)v.x; o[1] = (bf16)v.y; o[2] = (bf16)v.z; o[3] = (bf16)v.w;
  }
}

// ---------------------------------------------------------------- WMMA GEMM
// C[M,N] = A[M,K] @ W[N,K]^T.  Block tile 128(M) x 128(N) x 32(K), 8 waves,
// wave tile 32 x 64 (one full head wide). TDM double-buffered LDS staging.
// EPI: 0 = fp32 store, 1 = RMSNorm(lnw) + bf16 store, 2 = bf16 transposed
//      store into Vt[b, kv, hd, T].
template <int EPI>
__global__ __launch_bounds__(256) void gemm_kernel(
    const bf16* __restrict__ A, const bf16* __restrict__ W,
    float* __restrict__ Cf, bf16* __restrict__ Cb,
    const float* __restrict__ lnw, int M, int N, int K) {
  __shared__ bf16 As[2][128][40];   // 40-elem stride: conflict-free, 16B aligned
  __shared__ bf16 Ws[2][128][40];

  const int tid  = threadIdx.x;
  const int lane = tid & 31;
  const int wave = tid >> 5;
  const int wm   = wave & 3;   // 4 M-subtiles of 32
  const int wn   = wave >> 2;  // 2 N-subtiles of 64
  const int m0   = blockIdx.y * 128;
  const int n0   = blockIdx.x * 128;

  // prologue: stage k-slab 0 with the Tensor Data Mover
  if (wave == 0) {
    tdm_load_2d(A + (size_t)m0 * K, lds_of(&As[0][0][0]), 128, K);
    tdm_load_2d(W + (size_t)n0 * K, lds_of(&Ws[0][0][0]), 128, K);
    __builtin_amdgcn_s_wait_tensorcnt(0);
  }
  __syncthreads();

  const f32x8 z = {0.f,0.f,0.f,0.f,0.f,0.f,0.f,0.f};
  f32x8 acc[2][4];
#pragma unroll
  for (int mi = 0; mi < 2; ++mi)
#pragma unroll
    for (int j = 0; j < 4; ++j) acc[mi][j] = z;

  int buf = 0;
  for (int kt = 0; kt < K; kt += 32, buf ^= 1) {
    // issue next slab's TDM transfers; they overlap with this slab's WMMAs
    if (wave == 0 && kt + 32 < K) {
      tdm_load_2d(A + (size_t)m0 * K + (kt + 32), lds_of(&As[buf ^ 1][0][0]), 128, K);
      tdm_load_2d(W + (size_t)n0 * K + (kt + 32), lds_of(&Ws[buf ^ 1][0][0]), 128, K);
    }

    // A fragments: lane = M row; K 0-7/16-23 (lanes 0-15), 8-15/24-31 (16-31)
    Frag fa[2];
#pragma unroll
    for (int mi = 0; mi < 2; ++mi) {
      const int m  = wm * 32 + mi * 16 + (lane & 15);
      const int kb = (lane >> 4) * 8;
      fa[mi].h[0] = *(const bf16x8*)&As[buf][m][kb];
      fa[mi].h[1] = *(const bf16x8*)&As[buf][m][kb + 16];
    }
    // B fragments: lane = N col; K 0-7/16-23 (lanes 0-15), 8-15/24-31 (16-31)
#pragma unroll
    for (int j = 0; j < 4; ++j) {
      Frag fb;
      const int n   = wn * 64 + j * 16 + (lane & 15);
      const int kb2 = (lane >> 4) * 8;
      fb.h[0] = *(const bf16x8*)&Ws[buf][n][kb2];
      fb.h[1] = *(const bf16x8*)&Ws[buf][n][kb2 + 16];
#pragma unroll
      for (int mi = 0; mi < 2; ++mi) acc[mi][j] = wmma_bf16(fa[mi], fb, acc[mi][j]);
    }

    if (wave == 0 && kt + 32 < K) __builtin_amdgcn_s_wait_tensorcnt(0);
    __syncthreads();
  }

  // ----- epilogue (C layout: lanes 0-15 -> M=r, lanes 16-31 -> M=r+8; N=lane&15)
  const int nloc = lane & 15;
  const int mh   = (lane >> 4) << 3;

  if constexpr (EPI == 1) {
    float wln[4];
#pragma unroll
    for (int j = 0; j < 4; ++j) wln[j] = lnw[j * 16 + nloc];  // offset within head
#pragma unroll
    for (int mi = 0; mi < 2; ++mi) {
#pragma unroll
      for (int r = 0; r < 8; ++r) {
        float ss = 0.f;
#pragma unroll
        for (int j = 0; j < 4; ++j) ss += acc[mi][j][r] * acc[mi][j][r];
        ss = redsum16(ss);
        const float sc = rsqrtf(ss * (1.0f / HD) + EPS);
        const int row = m0 + wm * 32 + mi * 16 + r + mh;
#pragma unroll
        for (int j = 0; j < 4; ++j) {
          const int col = n0 + wn * 64 + j * 16 + nloc;
          Cb[(size_t)row * N + col] = (bf16)(wln[j] * acc[mi][j][r] * sc);
        }
      }
    }
  } else if constexpr (EPI == 2) {
#pragma unroll
    for (int mi = 0; mi < 2; ++mi) {
#pragma unroll
      for (int r = 0; r < 8; ++r) {
        const int row = m0 + wm * 32 + mi * 16 + r + mh;
        const int b = row >> 11, t = row & (T_SEQ - 1);
#pragma unroll
        for (int j = 0; j < 4; ++j) {
          const int col = n0 + wn * 64 + j * 16 + nloc;
          const int kv = col >> 6, d = col & 63;
          Cb[((size_t)((b * NKV + kv) * HD + d)) * T_SEQ + t] = (bf16)acc[mi][j][r];
        }
      }
    }
  } else {
#pragma unroll
    for (int mi = 0; mi < 2; ++mi) {
#pragma unroll
      for (int r = 0; r < 8; ++r) {
        const int row = m0 + wm * 32 + mi * 16 + r + mh;
#pragma unroll
        for (int j = 0; j < 4; ++j) {
          const int col = n0 + wn * 64 + j * 16 + nloc;
          Cf[(size_t)row * N + col] = acc[mi][j][r];
        }
      }
    }
  }
}

// ---------------------------------------------------------------- attention
// One wave per 16-query tile per head; online softmax over 32 chunks of 64
// keys. 16 WMMAs per chunk (8 QK^T + 8 PV).
__global__ __launch_bounds__(128) void attn_kernel(
    const bf16* __restrict__ Qb,  // [ROWS, DMODEL]  (post-RMSNorm)
    const bf16* __restrict__ Kb,  // [ROWS, KVD]     (post-RMSNorm)
    const bf16* __restrict__ Vt,  // [B, NKV, HD, T] (transposed)
    bf16* __restrict__ Ob) {      // [ROWS, DMODEL]
  __shared__ bf16 Pbuf[4][16][80];  // 80-elem stride: 2-way max bank conflict

  const int lane = threadIdx.x & 31;
  const int warp = threadIdx.x >> 5;
  const int w    = blockIdx.x * 4 + warp;        // 6144 waves total
  const int qt   = w & 127;                      // 128 query tiles
  const int rest = w >> 7;
  const int h    = rest % NH;
  const int b    = rest / NH;
  const int kv   = h / GRP;
  const int t0   = qt * 16;

  bf16 (*Pw)[80] = Pbuf[warp];

  const int nloc = lane & 15;
  const int mh   = (lane >> 4) << 3;

  // Q fragments (d 0..31 and 32..63)
  Frag aq0, aq1;
  {
    const bf16* qp = Qb + (size_t)(b * T_SEQ + t0 + nloc) * DMODEL + h * HD;
    const int kb = (lane >> 4) * 8;
    aq0.h[0] = *(const bf16x8*)(qp + kb);
    aq0.h[1] = *(const bf16x8*)(qp + kb + 16);
    aq1.h[0] = *(const bf16x8*)(qp + 32 + kb);
    aq1.h[1] = *(const bf16x8*)(qp + 32 + kb + 16);
  }

  const f32x8 z = {0.f,0.f,0.f,0.f,0.f,0.f,0.f,0.f};
  f32x8 accO[4] = {z, z, z, z};
  float mrow[8], lrow[8];
#pragma unroll
  for (int r = 0; r < 8; ++r) { mrow[r] = -__builtin_inff(); lrow[r] = 0.f; }

  for (int kc = 0; kc < T_SEQ; kc += 64) {
    // scores for 4 key sub-tiles of 16 (hd=64 -> two K-steps each)
    f32x8 S[4];
#pragma unroll
    for (int s = 0; s < 4; ++s) {
      Frag k0, k1;
      const bf16* kp = Kb + (size_t)(b * T_SEQ + kc + s * 16 + nloc) * KVD +
                       kv * HD + (lane >> 4) * 8;
      k0.h[0] = *(const bf16x8*)kp;        k0.h[1] = *(const bf16x8*)(kp + 16);
      k1.h[0] = *(const bf16x8*)(kp + 32); k1.h[1] = *(const bf16x8*)(kp + 48);
      S[s] = wmma_bf16(aq0, k0, z);
      S[s] = wmma_bf16(aq1, k1, S[s]);
    }

    // online softmax per row; stash P (bf16) into wave-private LDS
#pragma unroll
    for (int r = 0; r < 8; ++r) {
      float mx = fmaxf(fmaxf(S[0][r], S[1][r]), fmaxf(S[2][r], S[3][r]));
      mx = redmax16(mx);
      const float mnew  = fmaxf(mrow[r], mx);
      const float alpha = exp2f((mrow[r] - mnew) * SL2E);
      mrow[r] = mnew;
      const int m = r + mh;
      float ps = 0.f;
#pragma unroll
      for (int s = 0; s < 4; ++s) {
        const float p = exp2f((S[s][r] - mnew) * SL2E);
        ps += p;
        Pw[m][s * 16 + nloc] = (bf16)p;
      }
      ps = redsum16(ps);
      lrow[r] = lrow[r] * alpha + ps;
#pragma unroll
      for (int j = 0; j < 4; ++j) accO[j][r] *= alpha;
    }

    // P x V: two A-fragments (keys 0-31, 32-63) x four 16-wide d tiles
#pragma unroll
    for (int half = 0; half < 2; ++half) {
      Frag ap;
      const int kb = (lane >> 4) * 8 + half * 32;
      ap.h[0] = *(const bf16x8*)&Pw[nloc][kb];
      ap.h[1] = *(const bf16x8*)&Pw[nloc][kb + 16];
#pragma unroll
      for (int j = 0; j < 4; ++j) {
        Frag fv;
        const bf16* vp = Vt + (size_t)((b * NKV + kv) * HD + j * 16 + nloc) * T_SEQ +
                         kc + half * 32 + (lane >> 4) * 8;
        fv.h[0] = *(const bf16x8*)vp;
        fv.h[1] = *(const bf16x8*)(vp + 16);
        accO[j] = wmma_bf16(ap, fv, accO[j]);
      }
    }
  }

  // normalize and store
#pragma unroll
  for (int r = 0; r < 8; ++r) {
    const float inv = 1.0f / lrow[r];
    const int row = t0 + r + mh;
    bf16* op = Ob + (size_t)(b * T_SEQ + row) * DMODEL + h * HD;
#pragma unroll
    for (int j = 0; j < 4; ++j) op[j * 16 + nloc] = (bf16)(accO[j][r] * inv);
  }
}

// ---------------------------------------------------------------- launcher
extern "C" void kernel_launch(void* const* d_in, const int* in_sizes, int n_in,
                              void* d_out, int out_size, void* d_ws, size_t ws_size,
                              hipStream_t stream) {
  const float* x    = (const float*)d_in[0];
  const float* Wq   = (const float*)d_in[1];
  const float* Wk   = (const float*)d_in[2];
  const float* Wv   = (const float*)d_in[3];
  const float* Wo   = (const float*)d_in[4];
  const float* qlnw = (const float*)d_in[5];
  const float* klnw = (const float*)d_in[6];
  float* out = (float*)d_out;

  char* ws = (char*)d_ws;
  size_t off = 0;
  auto carve = [&](size_t bytes) { char* p = ws + off; off += (bytes + 255) & ~size_t(255); return p; };
  bf16* xb  = (bf16*)carve((size_t)ROWS * DMODEL * 2);
  bf16* wqb = (bf16*)carve((size_t)DMODEL * DMODEL * 2);
  bf16* wkb = (bf16*)carve((size_t)KVD * DMODEL * 2);
  bf16* wvb = (bf16*)carve((size_t)KVD * DMODEL * 2);
  bf16* wob = (bf16*)carve((size_t)DMODEL * DMODEL * 2);
  bf16* Qb  = (bf16*)carve((size_t)ROWS * DMODEL * 2);
  bf16* Kbb = (bf16*)carve((size_t)ROWS * KVD * 2);
  bf16* Vt  = (bf16*)carve((size_t)ROWS * KVD * 2);
  bf16* Ob  = (bf16*)carve((size_t)ROWS * DMODEL * 2);

  auto cvt = [&](const float* src, bf16* dst, int n) {
    int n4 = n / 4;
    cvt_kernel<<<(n4 + 255) / 256, 256, 0, stream>>>(src, dst, n4);
  };
  cvt(x,  xb,  ROWS * DMODEL);
  cvt(Wq, wqb, DMODEL * DMODEL);
  cvt(Wk, wkb, KVD * DMODEL);
  cvt(Wv, wvb, KVD * DMODEL);
  cvt(Wo, wob, DMODEL * DMODEL);

  // Q projection + fused per-head RMSNorm
  gemm_kernel<1><<<dim3(DMODEL / 128, ROWS / 128), 256, 0, stream>>>(
      xb, wqb, nullptr, Qb, qlnw, ROWS, DMODEL, DMODEL);
  // K projection + fused per-head RMSNorm
  gemm_kernel<1><<<dim3(KVD / 128, ROWS / 128), 256, 0, stream>>>(
      xb, wkb, nullptr, Kbb, klnw, ROWS, KVD, DMODEL);
  // V projection, stored transposed [b, kv, hd, T]
  gemm_kernel<2><<<dim3(KVD / 128, ROWS / 128), 256, 0, stream>>>(
      xb, wvb, nullptr, Vt, nullptr, ROWS, KVD, DMODEL);

  // flash attention: 6144 waves, 4 per block
  attn_kernel<<<(BATCH * NH * (T_SEQ / 16)) / 4, 128, 0, stream>>>(Qb, Kbb, Vt, Ob);

  // output projection straight into d_out (fp32)
  gemm_kernel<0><<<dim3(DMODEL / 128, ROWS / 128), 256, 0, stream>>>(
      Ob, wob, out, nullptr, nullptr, ROWS, DMODEL, DMODEL);
}
